// PPO_71708773974232
// MI455X (gfx1250) — matrix operand order repaired and checked by reading, not compiled, not run on hardware
//
#include <hip/hip_runtime.h>
#include <hip/hip_bf16.h>
#include <math.h>

#define TOK 65536
#define SD  128
#define HD  256
#define AD  8

#define GAMMA_F   0.99f
#define LMBDA_F   0.95f
#define LOG2PI_F  1.8378770664093453f

// LDS row strides (padded to break bank conflicts)
#define SA_STR 136   // halves; 272 B/row -> b128 groups shift 4 banks/row
#define H1_STR 264   // halves; 528 B/row
#define H2_STR 257   // floats; 1 bank shift/row -> conflict-free head reads

typedef __attribute__((ext_vector_type(16))) _Float16 v16h;
typedef __attribute__((ext_vector_type(8)))  _Float16 v8h;
typedef __attribute__((ext_vector_type(8)))  float    v8f;

// Branch-free tanh: tanh(x) = 1 - 2/(exp(2x)+1); v_exp_f32 saturates so the
// identity holds at +/-inf. ~1e-7 abs error, 5 VALU ops, no EXEC divergence.
__device__ __forceinline__ float fast_tanh(float x) {
  float e = __builtin_amdgcn_exp2f(x * 2.8853900817779268f); // 2*log2(e)
  return 1.0f - 2.0f * __builtin_amdgcn_rcpf(e + 1.0f);
}

// ---------------------------------------------------------------------------
// Weight packing: W[K x N] f32 row-major -> B-fragment-tiled f16.
// Block b = nt*(K/32)+kt is 512 halves; lane l (n=l&15, h=l>>4) holds
// halves j=0..15 = W[kt*32 + 16*h + j][nt*16 + n]  (contiguous k order, same as the A loads).
// ---------------------------------------------------------------------------
__global__ __launch_bounds__(256) void pack_weights(const float* __restrict__ W,
                                                    _Float16* __restrict__ Wp,
                                                    int KK, int NN) {
  int nK = KK / 32;
  int total = KK * NN;
  for (int idx = blockIdx.x * blockDim.x + threadIdx.x; idx < total;
       idx += gridDim.x * blockDim.x) {
    int blk = idx >> 9, rem = idx & 511;
    int l = rem >> 4, j = rem & 15;
    int h = l >> 4, n = l & 15;
    int kt = blk % nK, nt = blk / nK;
    Wp[idx] = (_Float16)W[(kt * 32 + 16 * h + j) * NN + nt * 16 + n];
  }
}

// ---------------------------------------------------------------------------
// Shared WMMA MLP core: 32 rows per workgroup, 128 threads (4 waves, wave32).
// ---------------------------------------------------------------------------
struct alignas(32) MLPSmem {
  _Float16 sA[32 * SA_STR];  // 8.5 KB  (f16 input tile)
  _Float16 h1[32 * H1_STR];  // 16.5 KB (layer-1 activations, f16)
  float    h2[32 * H2_STR];  // 32.1 KB (layer-2 activations, f32)
};

__device__ __forceinline__ v16h load_a_frag(const _Float16* rowp, int kt, int hsel) {
  v8h lo = *(const v8h*)(rowp + kt * 32 + 16 * hsel);
  v8h hi = *(const v8h*)(rowp + kt * 32 + 16 * hsel + 8);
  v16h f;
#pragma unroll
  for (int j = 0; j < 8; ++j) { f[j] = lo[j]; f[8 + j] = hi[j]; }
  return f;
}

__device__ __forceinline__ void mlp_two_layers(MLPSmem& sm,
                                               const float* __restrict__ x,
                                               const _Float16* __restrict__ W1p,
                                               const float* __restrict__ b1,
                                               const _Float16* __restrict__ W2p,
                                               const float* __restrict__ b2) {
  const int tid = threadIdx.x;
  for (int i = tid; i < 32 * SD; i += 128)
    sm.sA[(i >> 7) * SA_STR + (i & 127)] = (_Float16)x[i];
  __syncthreads();

  const int wave = tid >> 5;
  const int lane = tid & 31;
  const int m = lane & 15;        // A row within subtile / D column n
  const int hsel = lane >> 4;
  const int mt = wave >> 1;       // waves {0,1}->rows 0..15, {2,3}->16..31
  const int ntBase = (wave & 1) * 8;

  // ---- layer 1: [32x128] @ [128x256], K=128 -> 4 WMMA per tile ----
  v16h a1[4];
  {
    const _Float16* rp = &sm.sA[(mt * 16 + m) * SA_STR];
#pragma unroll
    for (int kt = 0; kt < 4; ++kt) a1[kt] = load_a_frag(rp, kt, hsel);
  }
#pragma unroll
  for (int i = 0; i < 8; ++i) {
    int nt = ntBase + i;
    v8f acc = {};
#pragma unroll
    for (int kt = 0; kt < 4; ++kt) {
      v16h b = *(const v16h*)(W1p + (size_t)(nt * 4 + kt) * 512 + lane * 16);
      acc = __builtin_amdgcn_wmma_f32_16x16x32_f16(false, a1[kt], false, b,
                                                   (short)0, acc, false, false);
    }
    float bias = b1[nt * 16 + m];
#pragma unroll
    for (int r = 0; r < 8; ++r) {
      float v = fast_tanh(acc[r] + bias);
      sm.h1[(mt * 16 + 8 * hsel + r) * H1_STR + nt * 16 + m] = (_Float16)v;
    }
  }
  __syncthreads();

  // ---- layer 2: [32x256] @ [256x256], K=256 -> 8 WMMA per tile ----
  v16h a2[8];
  {
    const _Float16* rp = &sm.h1[(mt * 16 + m) * H1_STR];
#pragma unroll
    for (int kt = 0; kt < 8; ++kt) a2[kt] = load_a_frag(rp, kt, hsel);
  }
#pragma unroll
  for (int i = 0; i < 8; ++i) {
    int nt = ntBase + i;
    v8f acc = {};
#pragma unroll
    for (int kt = 0; kt < 8; ++kt) {
      v16h b = *(const v16h*)(W2p + (size_t)(nt * 8 + kt) * 512 + lane * 16);
      acc = __builtin_amdgcn_wmma_f32_16x16x32_f16(false, a2[kt], false, b,
                                                   (short)0, acc, false, false);
    }
    float bias = b2[nt * 16 + m];
#pragma unroll
    for (int r = 0; r < 8; ++r)
      sm.h2[(mt * 16 + 8 * hsel + r) * H2_STR + nt * 16 + m] =
          fast_tanh(acc[r] + bias);
  }
  __syncthreads();
}

// ---------------------------------------------------------------------------
// Value network over all T rows -> values[T]
// ---------------------------------------------------------------------------
__global__ __launch_bounds__(128) void value_mlp_kernel(
    const float* __restrict__ s, const _Float16* __restrict__ W1p,
    const float* __restrict__ b1, const _Float16* __restrict__ W2p,
    const float* __restrict__ b2, const float* __restrict__ vhW,
    const float* __restrict__ vhb, float* __restrict__ values) {
  __shared__ MLPSmem sm;
  __shared__ float vred[4][32];
  const int row0 = blockIdx.x * 32;
  mlp_two_layers(sm, s + (size_t)row0 * SD, W1p, b1, W2p, b2);
  const int tid = threadIdx.x;
  const int row = tid & 31, seg = tid >> 5;
  float acc = 0.f;
  const float* hr = &sm.h2[row * H2_STR + seg * 64];
  const float* wr = vhW + seg * 64;
  for (int k = 0; k < 64; ++k) acc += hr[k] * wr[k];
  vred[seg][row] = acc;
  __syncthreads();
  if (tid < 32)
    values[row0 + tid] =
        vred[0][tid] + vred[1][tid] + vred[2][tid] + vred[3][tid] + vhb[0];
}

// ---------------------------------------------------------------------------
// Policy network over all T rows -> per-row new log_prob lp[T]
// ---------------------------------------------------------------------------
__global__ __launch_bounds__(128) void policy_mlp_kernel(
    const float* __restrict__ s, const _Float16* __restrict__ W1p,
    const float* __restrict__ b1, const _Float16* __restrict__ W2p,
    const float* __restrict__ b2, const float* __restrict__ muW,
    const float* __restrict__ mub, const float* __restrict__ logstd,
    const float* __restrict__ aAct, float* __restrict__ lp) {
  __shared__ MLPSmem sm;
  __shared__ float mred[4][32][AD];
  const int row0 = blockIdx.x * 32;
  mlp_two_layers(sm, s + (size_t)row0 * SD, W1p, b1, W2p, b2);
  const int tid = threadIdx.x;
  const int row = tid & 31, seg = tid >> 5;
  float mu[AD];
#pragma unroll
  for (int j = 0; j < AD; ++j) mu[j] = 0.f;
  const float* hr = &sm.h2[row * H2_STR + seg * 64];
  const float* wr = muW + seg * 64 * AD;
  for (int k = 0; k < 64; ++k) {
    float hv = hr[k];
#pragma unroll
    for (int j = 0; j < AD; ++j) mu[j] += hv * wr[k * AD + j];
  }
#pragma unroll
  for (int j = 0; j < AD; ++j) mred[seg][row][j] = mu[j];
  __syncthreads();
  if (tid < 32) {
    const float* arow = aAct + (size_t)(row0 + tid) * AD;
    float acc = 0.f;
#pragma unroll
    for (int j = 0; j < AD; ++j) {
      float muj = mub[j] + mred[0][tid][j] + mred[1][tid][j] +
                  mred[2][tid][j] + mred[3][tid][j];
      float ls = logstd[j];
      float z = (arow[j] - muj) * expf(-ls);
      acc += -0.5f * z * z - ls - 0.5f * LOG2PI_F;
    }
    lp[row0 + tid] = acc;
  }
}

// ---------------------------------------------------------------------------
// v(sp[-1]) in full f32 (single row, trivial cost) -> scal[0]
// ---------------------------------------------------------------------------
__global__ __launch_bounds__(256) void vlast_kernel(
    const float* __restrict__ sp, const float* __restrict__ vW1,
    const float* __restrict__ vb1, const float* __restrict__ vW2,
    const float* __restrict__ vb2, const float* __restrict__ vhW,
    const float* __restrict__ vhb, float* __restrict__ scal) {
  __shared__ float x[SD];
  __shared__ float h1[HD];
  __shared__ float red[HD];
  int tid = threadIdx.x;
  const float* xr = sp + (size_t)(TOK - 1) * SD;
  if (tid < SD) x[tid] = xr[tid];
  __syncthreads();
  float acc = vb1[tid];
  for (int k = 0; k < SD; ++k) acc += x[k] * vW1[k * HD + tid];
  h1[tid] = fast_tanh(acc);
  __syncthreads();
  acc = vb2[tid];
  for (int k = 0; k < HD; ++k) acc += h1[k] * vW2[k * HD + tid];
  red[tid] = fast_tanh(acc) * vhW[tid];
  __syncthreads();
  for (int sft = 128; sft > 0; sft >>= 1) {
    if (tid < sft) red[tid] += red[tid + sft];
    __syncthreads();
  }
  if (tid == 0) scal[0] = red[0] + vhb[0];
}

// ---------------------------------------------------------------------------
// GAE: reverse linear recurrence gae_t = delta_t + (g*l*m_t)*gae_{t+1}.
// 256 chunks of 256 in ONE block: per-chunk (a,b) affine pass, thread-0
// combine, replay with true carry. Writes adv[T], scal[1]=sum, scal[2]=sumsq.
// ---------------------------------------------------------------------------
__global__ __launch_bounds__(256) void gae_kernel(
    const float* __restrict__ values, const float* __restrict__ r,
    const float* __restrict__ dmask, float* __restrict__ adv,
    float* __restrict__ scal) {
  __shared__ float aC[256], bC[256], gIn[256];
  __shared__ float s1[256], s2[256];
  const int c = threadIdx.x;
  const int lo = c * 256, hi = lo + 255;
  const float vlast = scal[0];
  const float GL = GAMMA_F * LMBDA_F;

  float g = 0.f, ap = 1.f;
  for (int t = hi; t >= lo; --t) {
    float m = dmask[t];
    float nv = (t == TOK - 1) ? vlast : values[t + 1];
    float delta = r[t] + GAMMA_F * nv * m - values[t];
    float coef = GL * m;
    g = delta + coef * g;
    ap *= coef;
  }
  aC[c] = ap; bC[c] = g;
  __syncthreads();
  if (c == 0) {
    float gg = 0.f;
    for (int cc = 255; cc >= 0; --cc) { gIn[cc] = gg; gg = bC[cc] + aC[cc] * gg; }
  }
  __syncthreads();

  g = gIn[c];
  float ls1 = 0.f, ls2 = 0.f;
  for (int t = hi; t >= lo; --t) {
    float m = dmask[t];
    float nv = (t == TOK - 1) ? vlast : values[t + 1];
    float delta = r[t] + GAMMA_F * nv * m - values[t];
    g = delta + GL * m * g;
    adv[t] = g;
    ls1 += g; ls2 += g * g;
  }
  s1[c] = ls1; s2[c] = ls2;
  __syncthreads();
  for (int sft = 128; sft > 0; sft >>= 1) {
    if (c < sft) { s1[c] += s1[c + sft]; s2[c] += s2[c + sft]; }
    __syncthreads();
  }
  if (c == 0) { scal[1] = s1[0]; scal[2] = s2[0]; }
}

// ---------------------------------------------------------------------------
// Per-element PPO terms + per-block deterministic tree reduction.
// diff = v(s) - returns = -adv_unnorm, Huber is symmetric.
// ---------------------------------------------------------------------------
__global__ __launch_bounds__(256) void loss_partial_kernel(
    const float* __restrict__ adv, const float* __restrict__ lp,
    const float* __restrict__ lp_old, const float* __restrict__ scal,
    float* __restrict__ pgPart, float* __restrict__ vfPart) {
  __shared__ float pgs[256], vfs[256];
  const int tid = threadIdx.x;
  const int i = blockIdx.x * 256 + tid;
  const float mean = scal[1] / (float)TOK;
  float var = (scal[2] - (float)TOK * mean * mean) / (float)(TOK - 1);
  var = var > 0.f ? var : 0.f;
  const float inv = 1.0f / (sqrtf(var) + 1e-8f);

  float a = adv[i];
  float an = (a - mean) * inv;
  float ratio = expf(lp[i] - lp_old[i]);
  float rc = ratio < 0.8f ? 0.8f : (ratio > 1.2f ? 1.2f : ratio);
  float t1 = ratio * an, t2 = rc * an;
  pgs[tid] = -(t1 < t2 ? t1 : t2);
  float ad = fabsf(a);
  vfs[tid] = ad < 1.f ? 0.5f * a * a : ad - 0.5f;
  __syncthreads();
  for (int sft = 128; sft > 0; sft >>= 1) {
    if (tid < sft) { pgs[tid] += pgs[tid + sft]; vfs[tid] += vfs[tid + sft]; }
    __syncthreads();
  }
  if (tid == 0) { pgPart[blockIdx.x] = pgs[0]; vfPart[blockIdx.x] = vfs[0]; }
}

__global__ __launch_bounds__(256) void final_kernel(
    const float* __restrict__ pgPart, const float* __restrict__ vfPart,
    const float* __restrict__ logstd, float* __restrict__ out) {
  __shared__ float pg[256], vf[256];
  const int tid = threadIdx.x;
  pg[tid] = pgPart[tid]; vf[tid] = vfPart[tid];
  __syncthreads();
  for (int sft = 128; sft > 0; sft >>= 1) {
    if (tid < sft) { pg[tid] += pg[tid + sft]; vf[tid] += vf[tid + sft]; }
    __syncthreads();
  }
  if (tid == 0) {
    float lsm = 0.f;
    for (int j = 0; j < AD; ++j) lsm += logstd[j];
    lsm /= (float)AD;
    float entropy = 0.5f + 0.5f * LOG2PI_F + lsm;
    out[0] = pg[0] / (float)TOK + 0.5f * (vf[0] / (float)TOK) - 0.01f * entropy;
  }
}

// ---------------------------------------------------------------------------
extern "C" void kernel_launch(void* const* d_in, const int* in_sizes, int n_in,
                              void* d_out, int out_size, void* d_ws, size_t ws_size,
                              hipStream_t stream) {
  (void)in_sizes; (void)n_in; (void)out_size; (void)ws_size;
  const float* s      = (const float*)d_in[0];
  const float* aAct   = (const float*)d_in[1];
  const float* r      = (const float*)d_in[2];
  const float* sp     = (const float*)d_in[3];
  const float* lpOld  = (const float*)d_in[4];
  const float* dmask  = (const float*)d_in[5];
  const float* piW1   = (const float*)d_in[6];
  const float* pib1   = (const float*)d_in[7];
  const float* piW2   = (const float*)d_in[8];
  const float* pib2   = (const float*)d_in[9];
  const float* muW    = (const float*)d_in[10];
  const float* mub    = (const float*)d_in[11];
  const float* logstd = (const float*)d_in[12];
  const float* vW1    = (const float*)d_in[13];
  const float* vb1    = (const float*)d_in[14];
  const float* vW2    = (const float*)d_in[15];
  const float* vb2    = (const float*)d_in[16];
  const float* vhW    = (const float*)d_in[17];
  const float* vhb    = (const float*)d_in[18];

  float* ws      = (float*)d_ws;
  float* values  = ws;                 // T
  float* lp      = ws + TOK;           // T
  float* adv     = ws + 2 * TOK;       // T
  float* scal    = ws + 3 * TOK;       // [0]=vlast [1]=sumAdv [2]=sumSq
  float* pgPart  = scal + 16;          // 256
  float* vfPart  = pgPart + 256;       // 256
  _Float16* piW1p = (_Float16*)(vfPart + 256);   // 32-byte aligned
  _Float16* piW2p = piW1p + SD * HD;
  _Float16* vW1p  = piW2p + HD * HD;
  _Float16* vW2p  = vW1p + SD * HD;

  pack_weights<<<64, 256, 0, stream>>>(piW1, piW1p, SD, HD);
  pack_weights<<<256, 256, 0, stream>>>(piW2, piW2p, HD, HD);
  pack_weights<<<64, 256, 0, stream>>>(vW1, vW1p, SD, HD);
  pack_weights<<<256, 256, 0, stream>>>(vW2, vW2p, HD, HD);

  value_mlp_kernel<<<TOK / 32, 128, 0, stream>>>(s, vW1p, vb1, vW2p, vb2,
                                                 vhW, vhb, values);
  vlast_kernel<<<1, 256, 0, stream>>>(sp, vW1, vb1, vW2, vb2, vhW, vhb, scal);
  policy_mlp_kernel<<<TOK / 32, 128, 0, stream>>>(s, piW1p, pib1, piW2p, pib2,
                                                  muW, mub, logstd, aAct, lp);
  gae_kernel<<<1, 256, 0, stream>>>(values, r, dmask, adv, scal);
  loss_partial_kernel<<<TOK / 256, 256, 0, stream>>>(adv, lp, lpOld, scal,
                                                     pgPart, vfPart);
  final_kernel<<<1, 256, 0, stream>>>(pgPart, vfPart, logstd, (float*)d_out);
}
